// ProteinGN_23364622090308
// MI455X (gfx1250) — matrix operand order repaired and checked by reading, not compiled, not run on hardware
//
#include <hip/hip_runtime.h>
#include <hip/hip_fp16.h>
#include <math.h>

#define N_NODES 100000
#define N_EDGES 1600000
#define N_GRAPHS 200
#define NPG 500   // nodes per graph (constant by construction)

typedef __attribute__((ext_vector_type(16))) _Float16 v16h;
typedef __attribute__((ext_vector_type(8)))  _Float16 v8h;
typedef __attribute__((ext_vector_type(8)))  float    v8f;

// ---------------------------------------------------------------- utilities

__global__ void fill_zero(float* __restrict__ p, long long n) {
  long long i = (long long)blockIdx.x * blockDim.x + threadIdx.x;
  long long stride = (long long)gridDim.x * blockDim.x;
  for (; i < n; i += stride) p[i] = 0.f;
}

__global__ void count_deg(const int* __restrict__ rcv, const int* __restrict__ egr,
                          unsigned* __restrict__ cnt_in, unsigned* __restrict__ cnt_e) {
  unsigned i = blockIdx.x * blockDim.x + threadIdx.x;
  if (i >= N_EDGES) return;
  atomicAdd(&cnt_in[rcv[i]], 1u);
  atomicAdd(&cnt_e[egr[i]], 1u);
}

// Pack [W0;W1;W2] (f32, row-major [rows][OC]) into WMMA-B *fragment order* f16:
// dst[((s*NT + t)*32 + lane)*16 + j] = Wcat[s*32 + (lane>=16)*8 + (j&7) + (j>>3)*16][t*16 + (lane&15)]
__global__ void pack_frag(const float* __restrict__ W0, int r0,
                          const float* __restrict__ W1, int r1,
                          const float* __restrict__ W2, int r2,
                          int KS, int NT, int OC, _Float16* __restrict__ dst) {
  int idx = blockIdx.x * blockDim.x + threadIdx.x;
  if (idx >= KS * NT * 512) return;
  int j = idx & 15;
  int lane = (idx >> 4) & 31;
  int t = (idx >> 9) % NT;
  int s = idx / (NT * 512);
  int col = t * 16 + (lane & 15);
  int k = s * 32 + (lane >> 4) * 8 + (j & 7) + (j >> 3) * 16;
  float v = 0.f;
  if (k < r0)                v = W0[k * OC + col];
  else if (k < r0 + r1)      v = W1[(k - r0) * OC + col];
  else if (k < r0 + r1 + r2) v = W2[(k - r0 - r1) * OC + col];
  dst[idx] = (_Float16)v;
}

// ---------------------------------------------------------------- encoder

__global__ void enc_global(const float* __restrict__ bg, float* __restrict__ g) {
  int i = blockIdx.x * blockDim.x + threadIdx.x;
  if (i >= N_GRAPHS * 4) return;
  g[i] = fmaxf(bg[i & 3], 0.f);
}

__global__ void enc_edge(const float* __restrict__ x,
                         const float* __restrict__ W1, const float* __restrict__ b1,
                         const float* __restrict__ W2, const float* __restrict__ b2,
                         _Float16* __restrict__ out) {
  unsigned i = blockIdx.x * blockDim.x + threadIdx.x;
  if (i >= N_EDGES) return;
  float x0 = x[i * 2 + 0], x1 = x[i * 2 + 1];
  float h[4];
#pragma unroll
  for (int c = 0; c < 4; ++c) h[c] = fmaxf(x0 * W1[c] + x1 * W1[4 + c] + b1[c], 0.f);
#pragma unroll
  for (int c = 0; c < 8; ++c) {
    float a = b2[c];
#pragma unroll
    for (int k = 0; k < 4; ++k) a += h[k] * W2[k * 8 + c];
    out[i * 8 + c] = (_Float16)fmaxf(a, 0.f);
  }
}

__global__ void enc_node(const float* __restrict__ x,
                         const float* __restrict__ W1, const float* __restrict__ b1,
                         const float* __restrict__ W2, const float* __restrict__ b2,
                         _Float16* __restrict__ out) {
  unsigned i = blockIdx.x * blockDim.x + threadIdx.x;
  if (i >= N_NODES) return;
  float h[8];
#pragma unroll
  for (int c = 0; c < 8; ++c) h[c] = b1[c];
  for (int k = 0; k < 83; ++k) {
    float v = x[i * 83u + k];
#pragma unroll
    for (int c = 0; c < 8; ++c) h[c] += v * W1[k * 8 + c];
  }
#pragma unroll
  for (int c = 0; c < 8; ++c) h[c] = fmaxf(h[c], 0.f);
#pragma unroll
  for (int c = 0; c < 16; ++c) {
    float a = b2[c];
#pragma unroll
    for (int k = 0; k < 8; ++k) a += h[k] * W2[k * 16 + c];
    out[i * 16u + c] = (_Float16)fmaxf(a, 0.f);
  }
}

// ---------------------------------------------------------------- A-fragment 8-group loaders
// Segment boundaries (IE, IE+INN) are multiples of 8, and K0 is a multiple of 8
// after unrolling, so each group resolves at compile time to a single segment.

template <int IE, int INN, int IG>
__device__ __forceinline__ void load8e(v16h& a, int ofs, int K0,
                                       const _Float16* __restrict__ er,
                                       const _Float16* __restrict__ nr,
                                       const float* __restrict__ gr) {
  if (K0 + 8 <= IE) {                                   // fully in e (f16, b128)
    v8h v = *(const v8h*)(er + K0);
#pragma unroll
    for (int j = 0; j < 8; ++j) a[ofs + j] = v[j];
  } else if (K0 >= IE && K0 + 8 <= IE + INN) {          // fully in n[snd] (f16, b128)
    v8h v = *(const v8h*)(nr + (K0 - IE));
#pragma unroll
    for (int j = 0; j < 8; ++j) a[ofs + j] = v[j];
  } else if (K0 >= IE + INN && K0 + 8 <= IE + INN + IG) {  // fully in g (f32, 2xb128)
    float4 u = *(const float4*)(gr + (K0 - IE - INN));
    float4 w = *(const float4*)(gr + (K0 - IE - INN) + 4);
    a[ofs + 0] = (_Float16)u.x; a[ofs + 1] = (_Float16)u.y;
    a[ofs + 2] = (_Float16)u.z; a[ofs + 3] = (_Float16)u.w;
    a[ofs + 4] = (_Float16)w.x; a[ofs + 5] = (_Float16)w.y;
    a[ofs + 6] = (_Float16)w.z; a[ofs + 7] = (_Float16)w.w;
  } else {                                              // g/pad mixed group
#pragma unroll
    for (int j = 0; j < 8; ++j) {
      const int k = K0 + j;
      _Float16 v = (_Float16)0.f;
      if (k < IE)                      v = er[k];
      else if (k < IE + INN)           v = nr[k - IE];
      else if (k < IE + INN + IG)      v = (_Float16)gr[k - IE - INN];
      a[ofs + j] = v;
    }
  }
}

template <int INN, int OEIN, int IG>
__device__ __forceinline__ void load8n(v16h& a, int ofs, int K0,
                                       const _Float16* __restrict__ nr,
                                       const float* __restrict__ incr, float rinv,
                                       const float* __restrict__ gr) {
  if (K0 + 8 <= INN) {
    v8h v = *(const v8h*)(nr + K0);
#pragma unroll
    for (int j = 0; j < 8; ++j) a[ofs + j] = v[j];
  } else if (K0 >= INN && K0 + 8 <= INN + OEIN) {       // incoming mean (f32, 2xb128)
    float4 u = *(const float4*)(incr + (K0 - INN));
    float4 w = *(const float4*)(incr + (K0 - INN) + 4);
    a[ofs + 0] = (_Float16)(u.x * rinv); a[ofs + 1] = (_Float16)(u.y * rinv);
    a[ofs + 2] = (_Float16)(u.z * rinv); a[ofs + 3] = (_Float16)(u.w * rinv);
    a[ofs + 4] = (_Float16)(w.x * rinv); a[ofs + 5] = (_Float16)(w.y * rinv);
    a[ofs + 6] = (_Float16)(w.z * rinv); a[ofs + 7] = (_Float16)(w.w * rinv);
  } else if (K0 >= INN + OEIN && K0 + 8 <= INN + OEIN + IG) {
    float4 u = *(const float4*)(gr + (K0 - INN - OEIN));
    float4 w = *(const float4*)(gr + (K0 - INN - OEIN) + 4);
    a[ofs + 0] = (_Float16)u.x; a[ofs + 1] = (_Float16)u.y;
    a[ofs + 2] = (_Float16)u.z; a[ofs + 3] = (_Float16)u.w;
    a[ofs + 4] = (_Float16)w.x; a[ofs + 5] = (_Float16)w.y;
    a[ofs + 6] = (_Float16)w.z; a[ofs + 7] = (_Float16)w.w;
  } else {
#pragma unroll
    for (int j = 0; j < 8; ++j) {
      const int k = K0 + j;
      _Float16 v = (_Float16)0.f;
      if (k < INN)                       v = nr[k];
      else if (k < INN + OEIN)           v = (_Float16)(incr[k - INN] * rinv);
      else if (k < INN + OEIN + IG)      v = (_Float16)gr[k - INN - OEIN];
      a[ofs + j] = v;
    }
  }
}

// ---------------------------------------------------------------- WMMA edge update
template <int IE, int INN, int IG, int KP, int OE>
__global__ __launch_bounds__(256) void edge_update(
    const _Float16* __restrict__ e_in, const _Float16* __restrict__ n_in,
    const float* __restrict__ g_in, const _Float16* __restrict__ Wf,
    const float* __restrict__ be, const int* __restrict__ snd,
    const int* __restrict__ rcv, const int* __restrict__ egr,
    _Float16* __restrict__ e_out, float* __restrict__ inc_sum,
    float* __restrict__ geb_sum) {
  constexpr int NT = OE / 16;
  constexpr int KS = KP / 32;
  const int lane = threadIdx.x & 31;
  const int half = lane >> 4;
  const int mrow = lane & 15;
  const unsigned base = (blockIdx.x * 8u + (threadIdx.x >> 5)) * 16u;

  const unsigned aeg = base + mrow;
  const _Float16* er = e_in + aeg * IE;
  const _Float16* nr = n_in + (unsigned)snd[aeg] * INN;
  const float* gr = g_in + (unsigned)egr[aeg] * IG;

  // A fragments: ISA 16-bit A layout => elems 0-7 hold K=kb..kb+7,
  // elems 8-15 hold K=kb+16..kb+23, kb = 8*half. Build under half-wave branch
  // so every 8-group's segment is compile-time known; WMMA runs at full EXEC.
  v16h a[KS];
  if (half == 0) {
#pragma unroll
    for (int s = 0; s < KS; ++s) {
      load8e<IE, INN, IG>(a[s], 0, s * 32 + 0, er, nr, gr);
      load8e<IE, INN, IG>(a[s], 8, s * 32 + 16, er, nr, gr);
    }
  } else {
#pragma unroll
    for (int s = 0; s < KS; ++s) {
      load8e<IE, INN, IG>(a[s], 0, s * 32 + 8, er, nr, gr);
      load8e<IE, INN, IG>(a[s], 8, s * 32 + 24, er, nr, gr);
    }
  }

  v8f acc[NT];
#pragma unroll
  for (int t = 0; t < NT; ++t)
#pragma unroll
    for (int q = 0; q < 8; ++q) acc[t][q] = 0.f;

#pragma unroll
  for (int s = 0; s < KS; ++s)
#pragma unroll
    for (int t = 0; t < NT; ++t) {
      v16h b = *(const v16h*)(Wf + ((((unsigned)(s * NT + t)) * 32u + lane) << 4));
      acc[t] = __builtin_amdgcn_wmma_f32_16x16x32_f16(false, a[s], false, b,
                                                      (short)0, acc[t], false, false);
    }

  // epilogue: C/D layout -> lane holds rows m = 8*half + r, col = t*16 + mrow
  int rr[8], rg[8];
  {
    const int4* p4 = (const int4*)(rcv + base + half * 8);
    int4 r0 = p4[0], r1 = p4[1];
    rr[0] = r0.x; rr[1] = r0.y; rr[2] = r0.z; rr[3] = r0.w;
    rr[4] = r1.x; rr[5] = r1.y; rr[6] = r1.z; rr[7] = r1.w;
    const int4* q4 = (const int4*)(egr + base + half * 8);
    int4 g0 = q4[0], g1 = q4[1];
    rg[0] = g0.x; rg[1] = g0.y; rg[2] = g0.z; rg[3] = g0.w;
    rg[4] = g1.x; rg[5] = g1.y; rg[6] = g1.z; rg[7] = g1.w;
  }
#pragma unroll
  for (int t = 0; t < NT; ++t) {
    const int col = t * 16 + mrow;
    const float bias = be[col];
#pragma unroll
    for (int r = 0; r < 8; ++r) {
      const unsigned e = base + half * 8 + r;
      float x = fmaxf(acc[t][r] + bias, 0.f);
      e_out[e * OE + col] = (_Float16)x;
      unsafeAtomicAdd(inc_sum + ((unsigned)rr[r] * OE + col), x);
      unsafeAtomicAdd(geb_sum + ((unsigned)rg[r] * OE + col), x);
    }
  }
}

// ---------------------------------------------------------------- WMMA node update
template <int INN, int OEIN, int IG, int KP, int ON>
__global__ __launch_bounds__(256) void node_update(
    const _Float16* __restrict__ n_in, const float* __restrict__ inc_sum,
    const unsigned* __restrict__ cnt_in, const float* __restrict__ g_in,
    const _Float16* __restrict__ Wf, const float* __restrict__ bn,
    _Float16* __restrict__ n_out, float* __restrict__ gn_sum) {
  constexpr int NT = ON / 16;
  constexpr int KS = KP / 32;
  const int lane = threadIdx.x & 31;
  const int half = lane >> 4;
  const int mrow = lane & 15;
  const unsigned base = (blockIdx.x * 8u + (threadIdx.x >> 5)) * 16u;

  const unsigned arow = base + mrow;
  const unsigned nid = arow < N_NODES ? arow : (N_NODES - 1);
  const float rinv = 1.f / fmaxf((float)cnt_in[nid], 1.f);
  const _Float16* nr = n_in + nid * INN;
  const float* incr = inc_sum + nid * OEIN;
  const float* gr = g_in + (nid / NPG) * IG;

  v16h a[KS];
  if (half == 0) {
#pragma unroll
    for (int s = 0; s < KS; ++s) {
      load8n<INN, OEIN, IG>(a[s], 0, s * 32 + 0, nr, incr, rinv, gr);
      load8n<INN, OEIN, IG>(a[s], 8, s * 32 + 16, nr, incr, rinv, gr);
    }
  } else {
#pragma unroll
    for (int s = 0; s < KS; ++s) {
      load8n<INN, OEIN, IG>(a[s], 0, s * 32 + 8, nr, incr, rinv, gr);
      load8n<INN, OEIN, IG>(a[s], 8, s * 32 + 24, nr, incr, rinv, gr);
    }
  }

  v8f acc[NT];
#pragma unroll
  for (int t = 0; t < NT; ++t)
#pragma unroll
    for (int q = 0; q < 8; ++q) acc[t][q] = 0.f;

#pragma unroll
  for (int s = 0; s < KS; ++s)
#pragma unroll
    for (int t = 0; t < NT; ++t) {
      v16h b = *(const v16h*)(Wf + ((((unsigned)(s * NT + t)) * 32u + lane) << 4));
      acc[t] = __builtin_amdgcn_wmma_f32_16x16x32_f16(false, a[s], false, b,
                                                      (short)0, acc[t], false, false);
    }

#pragma unroll
  for (int t = 0; t < NT; ++t) {
    const int col = t * 16 + mrow;
    const float bias = bn[col];
#pragma unroll
    for (int r = 0; r < 8; ++r) {
      const unsigned node = base + half * 8 + r;
      if (node < N_NODES) {
        float x = fmaxf(acc[t][r] + bias, 0.f);
        n_out[node * ON + col] = (_Float16)x;
        unsafeAtomicAdd(gn_sum + ((node / NPG) * ON + col), x);
      }
    }
  }
}

// ---------------------------------------------------------------- global update (tiny)
template <int OE, int ON, int IG, int OG>
__global__ void global_update(const float* __restrict__ geb, const unsigned* __restrict__ cnt_e,
                              const float* __restrict__ gns, const float* __restrict__ g_in,
                              const float* __restrict__ Ge, const float* __restrict__ Gn,
                              const float* __restrict__ Gg, const float* __restrict__ bg,
                              float* __restrict__ g_out) {
  int idx = blockIdx.x * blockDim.x + threadIdx.x;
  if (idx >= N_GRAPHS * OG) return;
  int g = idx / OG, c = idx - g * OG;
  float re = 1.f / fmaxf((float)cnt_e[g], 1.f);
  constexpr float rn = 1.f / (float)NPG;
  float a = bg[c];
  for (int j = 0; j < OE; ++j) a += geb[g * OE + j] * re * Ge[j * OG + c];
  for (int j = 0; j < ON; ++j) a += gns[g * ON + j] * rn * Gn[j * OG + c];
  for (int j = 0; j < IG; ++j) a += g_in[g * IG + j] * Gg[j * OG + c];
  g_out[idx] = fmaxf(a, 0.f);
}

// ---------------------------------------------------------------- readout
__global__ void readout(const _Float16* __restrict__ n, const float* __restrict__ Wn,
                        const float* __restrict__ bn, const float* __restrict__ g,
                        const float* __restrict__ Wg, const float* __restrict__ bg,
                        float* __restrict__ out) {
  unsigned i = blockIdx.x * blockDim.x + threadIdx.x;
  if (i < N_NODES) {
    float a = bn[0];
    const v8h* p = (const v8h*)(n + i * 128u);
#pragma unroll
    for (int b = 0; b < 16; ++b) {
      v8h v = p[b];
#pragma unroll
      for (int j = 0; j < 8; ++j) a += (float)v[j] * Wn[b * 8 + j];
    }
    out[i] = 1.f / (1.f + __expf(-a));
  } else if (i < N_NODES + N_GRAPHS) {
    int gi = (int)(i - N_NODES);
    float a = bg[0];
    for (int k = 0; k < 32; ++k) a += g[gi * 32 + k] * Wg[k];
    out[i] = 1.f / (1.f + __expf(-a));
  }
}

// ---------------------------------------------------------------- launch

extern "C" void kernel_launch(void* const* d_in, const int* in_sizes, int n_in,
                              void* d_out, int out_size, void* d_ws, size_t ws_size,
                              hipStream_t stream) {
  (void)in_sizes; (void)n_in; (void)out_size; (void)ws_size;
  const float* node_features = (const float*)d_in[0];
  const float* edge_features = (const float*)d_in[1];
  const int* senders = (const int*)d_in[2];
  const int* receivers = (const int*)d_in[3];
  const int* edge_graph = (const int*)d_in[5];
  // encoder params (7..15)
  const float* We1 = (const float*)d_in[7];  const float* be1 = (const float*)d_in[8];
  const float* We2 = (const float*)d_in[9];  const float* be2 = (const float*)d_in[10];
  const float* Wn1 = (const float*)d_in[11]; const float* bn1 = (const float*)d_in[12];
  const float* Wn2 = (const float*)d_in[13]; const float* bn2 = (const float*)d_in[14];
  const float* enc_bg = (const float*)d_in[15];
  // mp layer params: 12 arrays each, order We,Ws,Weg,be,Wn,Win,Wng,bn,Ge,Gn,Gg,bg
  const float* L[3][12];
  for (int l = 0; l < 3; ++l)
    for (int j = 0; j < 12; ++j) L[l][j] = (const float*)d_in[16 + 12 * l + j];
  const float* roWn = (const float*)d_in[52]; const float* robn = (const float*)d_in[53];
  const float* roWg = (const float*)d_in[54]; const float* robg = (const float*)d_in[55];

  // ---- workspace carve (chunk sizes multiples of 256B so carve stays contiguous)
  char* w = (char*)d_ws;
  size_t off = 0;
  auto carve = [&](size_t bytes) -> char* {
    char* p = w + off;
    off = (off + bytes + 255) & ~(size_t)255;
    return p;
  };
  _Float16* eA = (_Float16*)carve((size_t)N_EDGES * 32 * 2);   // edge feats (<=32 wide)
  _Float16* eB = (_Float16*)carve((size_t)N_EDGES * 64 * 2);   // edge feats (<=64 wide)
  _Float16* nA = (_Float16*)carve((size_t)N_NODES * 64 * 2);   // node feats (<=64 wide)
  _Float16* nB = (_Float16*)carve((size_t)N_NODES * 128 * 2);  // node feats (<=128 wide)
  float* gA = (float*)carve((size_t)N_GRAPHS * 16 * 4);
  float* gB = (float*)carve((size_t)N_GRAPHS * 32 * 4);
  float* inc = (float*)carve((size_t)N_NODES * 64 * 4);        // incoming-edge sums
  float* geb = (float*)carve((size_t)N_GRAPHS * 64 * 4);       // per-graph edge sums
  float* gns = (float*)carve((size_t)N_GRAPHS * 128 * 4);      // per-graph node sums
  unsigned* cnt = (unsigned*)carve((size_t)(N_NODES + 256) * 4);
  unsigned* cnt_in = cnt;
  unsigned* cnt_e = cnt + N_NODES;
  _Float16* WE[3]; _Float16* WN[3];
  for (int l = 0; l < 3; ++l) WE[l] = (_Float16*)carve(128 * 64 * 2);
  for (int l = 0; l < 3; ++l) WN[l] = (_Float16*)carve(160 * 128 * 2);

  const long long FILLN = (long long)N_NODES * 64 + N_GRAPHS * 64 + N_GRAPHS * 128;
  const int EB = (N_EDGES + 255) / 256;
  const int NBK = (N_NODES + 255) / 256;

  // ---- degree counts (constant across layers)
  fill_zero<<<256, 256, 0, stream>>>((float*)cnt, N_NODES + 256);
  count_deg<<<EB, 256, 0, stream>>>(receivers, edge_graph, cnt_in, cnt_e);

  // ---- weight packing into B-fragment order: edge KS/NT = 1/1, 2/2, 4/4; node 2/2, 3/4, 5/8
  pack_frag<<<2, 256, 0, stream>>>(L[0][0], 8, L[0][1], 16, L[0][2], 4, 1, 1, 16, WE[0]);
  pack_frag<<<8, 256, 0, stream>>>(L[1][0], 16, L[1][1], 32, L[1][2], 8, 2, 2, 32, WE[1]);
  pack_frag<<<32, 256, 0, stream>>>(L[2][0], 32, L[2][1], 64, L[2][2], 16, 4, 4, 64, WE[2]);
  pack_frag<<<8, 256, 0, stream>>>(L[0][4], 16, L[0][5], 16, L[0][6], 4, 2, 2, 32, WN[0]);
  pack_frag<<<24, 256, 0, stream>>>(L[1][4], 32, L[1][5], 32, L[1][6], 8, 3, 4, 64, WN[1]);
  pack_frag<<<80, 256, 0, stream>>>(L[2][4], 64, L[2][5], 64, L[2][6], 16, 5, 8, 128, WN[2]);

  // ---- encoder
  enc_global<<<4, 256, 0, stream>>>(enc_bg, gA);
  enc_edge<<<EB, 256, 0, stream>>>(edge_features, We1, be1, We2, be2, eA);
  enc_node<<<NBK, 256, 0, stream>>>(node_features, Wn1, bn1, Wn2, bn2, nA);

  const int EWG = N_EDGES / 128;          // 12500 (exact)
  const int NWG = (N_NODES + 127) / 128;  // 782

  // ---- layer 1: (8,16,4) -> (16,32,8); e: A->B, n: A->B, g: A->B
  fill_zero<<<2048, 256, 0, stream>>>(inc, FILLN);
  edge_update<8, 16, 4, 32, 16><<<EWG, 256, 0, stream>>>(
      eA, nA, gA, WE[0], L[0][3], senders, receivers, edge_graph, eB, inc, geb);
  node_update<16, 16, 4, 64, 32><<<NWG, 256, 0, stream>>>(
      nA, inc, cnt_in, gA, WN[0], L[0][7], nB, gns);
  global_update<16, 32, 4, 8><<<(N_GRAPHS * 8 + 255) / 256, 256, 0, stream>>>(
      geb, cnt_e, gns, gA, L[0][8], L[0][9], L[0][10], L[0][11], gB);

  // ---- layer 2: (16,32,8) -> (32,64,16); e: B->A, n: B->A, g: B->A
  fill_zero<<<2048, 256, 0, stream>>>(inc, FILLN);
  edge_update<16, 32, 8, 64, 32><<<EWG, 256, 0, stream>>>(
      eB, nB, gB, WE[1], L[1][3], senders, receivers, edge_graph, eA, inc, geb);
  node_update<32, 32, 8, 96, 64><<<NWG, 256, 0, stream>>>(
      nB, inc, cnt_in, gB, WN[1], L[1][7], nA, gns);
  global_update<32, 64, 8, 16><<<(N_GRAPHS * 16 + 255) / 256, 256, 0, stream>>>(
      geb, cnt_e, gns, gB, L[1][8], L[1][9], L[1][10], L[1][11], gA);

  // ---- layer 3: (32,64,16) -> (64,128,32); e: A->B, n: A->B, g: A->B
  fill_zero<<<2048, 256, 0, stream>>>(inc, FILLN);
  edge_update<32, 64, 16, 128, 64><<<EWG, 256, 0, stream>>>(
      eA, nA, gA, WE[2], L[2][3], senders, receivers, edge_graph, eB, inc, geb);
  node_update<64, 64, 16, 160, 128><<<NWG, 256, 0, stream>>>(
      nA, inc, cnt_in, gA, WN[2], L[2][7], nB, gns);
  global_update<64, 128, 16, 32><<<(N_GRAPHS * 32 + 255) / 256, 256, 0, stream>>>(
      geb, cnt_e, gns, gA, L[2][8], L[2][9], L[2][10], L[2][11], gB);

  // ---- readout: node_out (100000) ++ global_out (200)
  readout<<<(N_NODES + N_GRAPHS + 255) / 256, 256, 0, stream>>>(
      nB, roWn, robn, gB, roWg, robg, (float*)d_out);
}
